// BiMambaBlock_41300405518476
// MI455X (gfx1250) — matrix operand order repaired and checked by reading, not compiled, not run on hardware
//
#include <hip/hip_runtime.h>
#include <math.h>

// Problem constants (B=2, L=1024, D_MODEL=1024, D_STATE=16, D_CONV=4, EXPAND=2)
#define TSZ    2048   // B*L tokens
#define NBATCH 2
#define LL     1024
#define DM     1024
#define DI     2048   // D_INNER
#define DSTATE 16
#define DTRANK 64
#define RCOLS  96     // DTRANK + 2*D_STATE
#define NXZ    4096   // 2*D_INNER

typedef __attribute__((ext_vector_type(16))) __bf16        v16bf;
typedef __attribute__((ext_vector_type(8)))  float         v8f;
typedef __attribute__((ext_vector_type(4)))  unsigned int  v4u;

union ABReg { v4u u[2]; v16bf v; };

__device__ __forceinline__ unsigned short f2bf(float f) {
  unsigned int u = __float_as_uint(f);
  u += 0x7FFFu + ((u >> 16) & 1u);   // round-to-nearest-even
  return (unsigned short)(u >> 16);
}

__global__ void f32_to_bf16_kernel(const float* __restrict__ src,
                                   unsigned short* __restrict__ dst, int n) {
  int i = blockIdx.x * blockDim.x + threadIdx.x;
  if (i < n) dst[i] = f2bf(src[i]);
}

// LayerNorm over rows of width DM; one block (256 thr) per row; bf16 output.
__global__ void ln_kernel(const float* __restrict__ x, const float* __restrict__ g,
                          const float* __restrict__ b, unsigned short* __restrict__ out_bf) {
  __shared__ float s1[256], s2[256];
  int row = blockIdx.x;
  const float* xr = x + (size_t)row * DM;
  float sum = 0.f, sq = 0.f;
  for (int i = threadIdx.x; i < DM; i += 256) { float v = xr[i]; sum += v; sq += v * v; }
  s1[threadIdx.x] = sum; s2[threadIdx.x] = sq;
  __syncthreads();
  for (int o = 128; o > 0; o >>= 1) {
    if (threadIdx.x < o) { s1[threadIdx.x] += s1[threadIdx.x + o]; s2[threadIdx.x] += s2[threadIdx.x + o]; }
    __syncthreads();
  }
  float mean = s1[0] * (1.f / DM);
  float var  = s2[0] * (1.f / DM) - mean * mean;
  float rstd = rsqrtf(var + 1e-5f);
  for (int i = threadIdx.x; i < DM; i += 256) {
    float v = (xr[i] - mean) * rstd * g[i] + b[i];
    out_bf[(size_t)row * DM + i] = f2bf(v);
  }
}

// ---------------------------------------------------------------------------
// Async LDS copy (CDNA5): one 16-byte global->LDS transfer per lane, ASYNCcnt.
__device__ __forceinline__ void async_cp16(unsigned int lds_off, const unsigned short* g) {
  asm volatile("global_load_async_to_lds_b128 %0, %1, off"
               :: "v"(lds_off), "v"(g) : "memory");
}
__device__ __forceinline__ void wait_async0() {
  asm volatile("s_wait_asynccnt 0x0" ::: "memory");
}

// ---------------------------------------------------------------------------
// LDS-tiled GEMM: C[M,N] = A[M,K] * W[N,K]^T, bf16 in, fp32 accum.
// Block = 256 thr (8 waves) computes a 128x128 tile; K consumed in 32-wide
// slices that are double-buffered in LDS via async copies. Each wave holds
// 8 16x16 accumulators (two groups of 4 to cap live VGPRs).
// EPI: 0=store f32; 1=softplus(x+bias) f32; 2=gelu(x+bias)->bf16; 3=resid+x+bias f32.
#define BM 128
#define BN 128
#define BK 32
#define ASTRIDE 40                    // row stride in elements (80B: 20 banks)
#define A_ELEMS (BM * ASTRIDE)        // 5120
#define B_ELEMS (BN * ASTRIDE)        // 5120

template<int EPI>
__global__ void gemm_tiled(const unsigned short* __restrict__ A, int lda,
                           const unsigned short* __restrict__ W,
                           int M, int N, int K,
                           const float* __restrict__ bias,
                           const float* __restrict__ resid,
                           float* __restrict__ outF,
                           unsigned short* __restrict__ outB) {
  __shared__ __align__(16) unsigned short smem[2][A_ELEMS + B_ELEMS];
  const int tid  = threadIdx.x;
  const int wid  = tid >> 5;
  const int lane = tid & 31;
  const int nTilesN = N / BN;
  const int rowBase = (blockIdx.x / nTilesN) * BM;
  const int colBase = (blockIdx.x % nTilesN) * BN;

  const int lr = tid >> 2;            // 0..63 : row handled by this loader thread
  const int lc = tid & 3;             // 0..3  : 16B chunk within a 64B row slice

  auto stage_load = [&](int s, int kb) {
    unsigned int base = (unsigned int)(size_t)(&smem[s][0]);
    // A slice: 128 rows x 32 elems (two passes of 64 rows)
#pragma unroll
    for (int p = 0; p < 2; ++p) {
      int r = p * 64 + lr;
      const unsigned short* g = A + (size_t)(rowBase + r) * lda + kb + lc * 8;
      async_cp16(base + (unsigned int)(r * ASTRIDE + lc * 8) * 2u, g);
    }
    // B slice: 128 rows (output cols) x 32 elems (two passes of 64 rows)
#pragma unroll
    for (int p = 0; p < 2; ++p) {
      int r = p * 64 + lr;
      const unsigned short* gb = W + (size_t)(colBase + r) * K + kb + lc * 8;
      async_cp16(base + (unsigned int)(A_ELEMS + r * ASTRIDE + lc * 8) * 2u, gb);
    }
  };

  const int m  = lane & 15;
  const int hi = lane >> 4;
  v8f acc[8];
  v8f zero = {};
#pragma unroll
  for (int t = 0; t < 8; ++t) acc[t] = zero;

  stage_load(0, 0);
  int cur = 0;
  for (int kb = 0; kb < K; kb += BK) {
    wait_async0();       // own async copies into buffer `cur` done
    __syncthreads();     // everyone's copies done AND everyone finished reading cur^1
    if (kb + BK < K) stage_load(cur ^ 1, kb + BK);
    const unsigned short* sa = &smem[cur][0];
    const unsigned short* sb = &smem[cur][A_ELEMS];
    // A fragment (16x32): lanes 0-15 K {0..7,16..23}, lanes 16-31 K {8..15,24..31}
    ABReg a;
    const unsigned short* ar = sa + (wid * 16 + m) * ASTRIDE + hi * 8;
    a.u[0] = *reinterpret_cast<const v4u*>(ar);
    a.u[1] = *reinterpret_cast<const v4u*>(ar + 16);
#pragma unroll
    for (int g = 0; g < 2; ++g) {
      // Load all four B fragments of this group first, then issue the WMMAs
      // back-to-back (single dscnt wait covers the group).
      ABReg bb[4];
#pragma unroll
      for (int t = 0; t < 4; ++t) {
        const unsigned short* br = sb + ((g * 4 + t) * 16 + m) * ASTRIDE + hi * 8;
        bb[t].u[0] = *reinterpret_cast<const v4u*>(br);
        bb[t].u[1] = *reinterpret_cast<const v4u*>(br + 16);
      }
#pragma unroll
      for (int t = 0; t < 4; ++t)
        acc[g * 4 + t] = __builtin_amdgcn_wmma_f32_16x16x32_bf16(
            false, a.v, false, bb[t].v, (short)0, acc[g * 4 + t], false, false);
    }
    cur ^= 1;
  }
  // C/D layout: lane col = lane&15; VGPR v holds row (lane>>4)*8 + v
#pragma unroll
  for (int t = 0; t < 8; ++t) {
    int n = colBase + t * 16 + m;
    float bi = 0.f;
    if constexpr (EPI >= 1) bi = bias[n];
#pragma unroll
    for (int v = 0; v < 8; ++v) {
      int row = rowBase + wid * 16 + hi * 8 + v;
      float val = acc[t][v];
      size_t oi = (size_t)row * N + n;
      if constexpr (EPI == 0) {
        outF[oi] = val;
      } else if constexpr (EPI == 1) {          // softplus
        float s = val + bi;
        outF[oi] = (s > 20.f) ? s : log1pf(expf(s));
      } else if constexpr (EPI == 2) {          // exact GELU -> bf16
        float s = val + bi;
        float gl = 0.5f * s * (1.f + erff(s * 0.70710678118654752f));
        outB[oi] = f2bf(gl);
      } else {                                  // residual + bias
        outF[oi] = resid[oi] + val + bi;
      }
    }
  }
}

// ---------------------------------------------------------------------------
// Simple register-direct WMMA GEMM (used for N=96 x_proj). One wave = 16x(16*NT).
template<int NT, int EPI>
__global__ void gemm_wmma_bf16(const unsigned short* __restrict__ A, int lda,
                               const unsigned short* __restrict__ W,
                               int M, int N, int K,
                               const float* __restrict__ bias,
                               const float* __restrict__ resid,
                               float* __restrict__ outF,
                               unsigned short* __restrict__ outB) {
  const int lane = threadIdx.x & 31;
  const int gw   = (blockIdx.x * blockDim.x + threadIdx.x) >> 5;
  const int wavesPerRow = N / (16 * NT);
  if (gw >= (M / 16) * wavesPerRow) return;
  const int tileM = gw / wavesPerRow;
  const int nBase = (gw % wavesPerRow) * (16 * NT);
  const int m  = lane & 15;
  const int hi = lane >> 4;
  const unsigned short* arow = A + (size_t)(tileM * 16 + m) * lda + hi * 8;
  const int ks = hi * 8;
  v8f acc[NT];
  v8f zero = {};
#pragma unroll
  for (int t = 0; t < NT; ++t) acc[t] = zero;
  for (int kb = 0; kb < K; kb += 32) {
    ABReg a;
    a.u[0] = *reinterpret_cast<const v4u*>(arow + kb);
    a.u[1] = *reinterpret_cast<const v4u*>(arow + kb + 16);
    ABReg bb[NT];
#pragma unroll
    for (int t = 0; t < NT; ++t) {
      const unsigned short* wrow = W + (size_t)(nBase + t * 16 + m) * K + kb + ks;
      bb[t].u[0] = reinterpret_cast<const v4u*>(wrow)[0];
      bb[t].u[1] = reinterpret_cast<const v4u*>(wrow)[2];
    }
#pragma unroll
    for (int t = 0; t < NT; ++t)
      acc[t] = __builtin_amdgcn_wmma_f32_16x16x32_bf16(false, a.v, false, bb[t].v,
                                                       (short)0, acc[t], false, false);
  }
#pragma unroll
  for (int t = 0; t < NT; ++t) {
    int n = nBase + t * 16 + m;
    float bi = 0.f;
    if constexpr (EPI >= 1) bi = bias[n];
#pragma unroll
    for (int v = 0; v < 8; ++v) {
      int row = tileM * 16 + hi * 8 + v;
      float val = acc[t][v];
      size_t oi = (size_t)row * N + n;
      if constexpr (EPI == 0) {
        outF[oi] = val;
      } else if constexpr (EPI == 1) {
        float s = val + bi;
        outF[oi] = (s > 20.f) ? s : log1pf(expf(s));
      } else if constexpr (EPI == 2) {
        float s = val + bi;
        float gl = 0.5f * s * (1.f + erff(s * 0.70710678118654752f));
        outB[oi] = f2bf(gl);
      } else {
        outF[oi] = resid[oi] + val + bi;
      }
    }
  }
}

// Depthwise causal conv (k=4) + bias + SiLU. rev=1 reads the flipped sequence.
__global__ void conv_silu_kernel(const float* __restrict__ xz, const float* __restrict__ w,
                                 const float* __restrict__ bias, float* __restrict__ u,
                                 unsigned short* __restrict__ u_bf, int rev) {
  int idx = blockIdx.x * blockDim.x + threadIdx.x;
  if (idx >= TSZ * DI) return;
  int c = idx % DI;
  int t = idx / DI;
  int bidx = t / LL, l = t % LL;
  float acc = bias[c];
#pragma unroll
  for (int j = 0; j < 4; ++j) {
    int lp = l - 3 + j;
    if (lp >= 0) {
      int ls = rev ? (LL - 1 - lp) : lp;
      acc += w[c * 4 + j] * xz[((size_t)(bidx * LL + ls)) * NXZ + c];
    }
  }
  float s = acc / (1.f + expf(-acc));
  u[idx] = s;
  u_bf[idx] = f2bf(s);
}

// Selective scan: 16 lanes per channel (one per state n); sequential over L.
// Software-pipelined: loads for step l+1 are issued before computing step l.
__global__ void scan_kernel(const float* __restrict__ u, const float* __restrict__ dt,
                            const float* __restrict__ xdbl, const float* __restrict__ A_log,
                            float* __restrict__ yscan) {
  int gid  = blockIdx.x * blockDim.x + threadIdx.x;
  int lane = gid & 31;
  int wave = gid >> 5;
  int chan = wave * 2 + (lane >> 4);
  if (chan >= NBATCH * DI) return;
  int n = lane & 15;
  int bidx = chan / DI;
  int d = chan % DI;
  float An = -expf(A_log[d * DSTATE + n]);
  float h = 0.f;
  size_t tbase = (size_t)bidx * LL;
  float dt_c = dt[tbase * DI + d];
  float u_c  = u[tbase * DI + d];
  float B_c  = xdbl[tbase * RCOLS + DTRANK + n];
  float C_c  = xdbl[tbase * RCOLS + DTRANK + DSTATE + n];
  for (int l = 0; l < LL; ++l) {
    size_t t = tbase + l;
    float dt_nx = 0.f, u_nx = 0.f, B_nx = 0.f, C_nx = 0.f;
    if (l + 1 < LL) {                 // prefetch next step while we compute
      size_t tn = t + 1;
      dt_nx = dt[tn * DI + d];
      u_nx  = u[tn * DI + d];
      B_nx  = xdbl[tn * RCOLS + DTRANK + n];
      C_nx  = xdbl[tn * RCOLS + DTRANK + DSTATE + n];
    }
    h = expf(dt_c * An) * h + (dt_c * u_c) * B_c;
    float yp = h * C_c;
    yp += __shfl_xor(yp, 8, 32);
    yp += __shfl_xor(yp, 4, 32);
    yp += __shfl_xor(yp, 2, 32);
    yp += __shfl_xor(yp, 1, 32);
    if (n == 0) yscan[t * DI + d] = yp;
    dt_c = dt_nx; u_c = u_nx; B_c = B_nx; C_c = C_nx;
  }
}

// y = (scan + u*D) * silu(z); z read at flipped position for rev=1.
__global__ void gate_kernel(const float* __restrict__ yscan, const float* __restrict__ u,
                            const float* __restrict__ xz, const float* __restrict__ Dp,
                            unsigned short* __restrict__ y_bf, int rev) {
  int idx = blockIdx.x * blockDim.x + threadIdx.x;
  if (idx >= TSZ * DI) return;
  int c = idx % DI, t = idx / DI;
  int bidx = t / LL, l = t % LL;
  int zt = rev ? (bidx * LL + (LL - 1 - l)) : t;
  float z = xz[(size_t)zt * NXZ + DI + c];
  float y = (yscan[idx] + u[idx] * Dp[c]) * (z / (1.f + expf(-z)));
  y_bf[idx] = f2bf(y);
}

// x2 = x + 0.5*(y_fwd + flip(y_bwd))
__global__ void combine_kernel(const float* __restrict__ x, const float* __restrict__ yf,
                               const float* __restrict__ yb, float* __restrict__ x2) {
  int idx = blockIdx.x * blockDim.x + threadIdx.x;
  if (idx >= TSZ * DM) return;
  int dm = idx % DM, t = idx / DM;
  int bidx = t / LL, l = t % LL;
  size_t tb = (size_t)(bidx * LL + (LL - 1 - l));
  x2[idx] = x[idx] + 0.5f * (yf[idx] + yb[tb * DM + dm]);
}

extern "C" void kernel_launch(void* const* d_in, const int* in_sizes, int n_in,
                              void* d_out, int out_size, void* d_ws, size_t ws_size,
                              hipStream_t stream) {
  (void)in_sizes; (void)n_in; (void)out_size; (void)ws_size;
  const float* x      = (const float*)d_in[0];
  const float* norm_g = (const float*)d_in[1];
  const float* norm_b = (const float*)d_in[2];
  const float* in_w[2]   = { (const float*)d_in[3],  (const float*)d_in[12] };
  const float* conv_w[2] = { (const float*)d_in[4],  (const float*)d_in[13] };
  const float* conv_b[2] = { (const float*)d_in[5],  (const float*)d_in[14] };
  const float* xp_w[2]   = { (const float*)d_in[6],  (const float*)d_in[15] };
  const float* dt_w[2]   = { (const float*)d_in[7],  (const float*)d_in[16] };
  const float* dt_bias[2]= { (const float*)d_in[8],  (const float*)d_in[17] };
  const float* A_log[2]  = { (const float*)d_in[9],  (const float*)d_in[18] };
  const float* Dp[2]     = { (const float*)d_in[10], (const float*)d_in[19] };
  const float* out_w[2]  = { (const float*)d_in[11], (const float*)d_in[20] };
  const float* ffn_g = (const float*)d_in[21];
  const float* ffn_b = (const float*)d_in[22];
  const float* w1 = (const float*)d_in[23];
  const float* b1 = (const float*)d_in[24];
  const float* w2 = (const float*)d_in[25];
  const float* b2 = (const float*)d_in[26];

  char* ws = (char*)d_ws;
  size_t off = 0;
  auto alloc = [&](size_t bytes) -> char* {
    char* p = ws + off;
    off = (off + bytes + 255) & ~(size_t)255;
    return p;
  };
  unsigned short* xn_bf = (unsigned short*)alloc((size_t)TSZ * DM * 2);
  unsigned short *in_wb[2], *xp_wb[2], *dt_wb[2], *out_wb[2];
  for (int d = 0; d < 2; ++d) {
    in_wb[d]  = (unsigned short*)alloc((size_t)NXZ * DM * 2);
    xp_wb[d]  = (unsigned short*)alloc((size_t)RCOLS * DI * 2);
    dt_wb[d]  = (unsigned short*)alloc((size_t)DI * DTRANK * 2);
    out_wb[d] = (unsigned short*)alloc((size_t)DM * DI * 2);
  }
  unsigned short* w1b = (unsigned short*)alloc((size_t)4096 * DM * 2);
  unsigned short* w2b = (unsigned short*)alloc((size_t)DM * 4096 * 2);
  float* xz    = (float*)alloc((size_t)TSZ * NXZ * 4);
  float* u     = (float*)alloc((size_t)TSZ * DI * 4);
  unsigned short* u_bf = (unsigned short*)alloc((size_t)TSZ * DI * 2);
  float* xdbl  = (float*)alloc((size_t)TSZ * RCOLS * 4);
  unsigned short* xdbl_bf = (unsigned short*)alloc((size_t)TSZ * RCOLS * 2);
  float* dtb   = (float*)alloc((size_t)TSZ * DI * 4);
  float* yscan = (float*)alloc((size_t)TSZ * DI * 4);
  unsigned short* y_bf = (unsigned short*)alloc((size_t)TSZ * DI * 2);
  float* yout[2] = { (float*)alloc((size_t)TSZ * DM * 4), (float*)alloc((size_t)TSZ * DM * 4) };
  float* x2 = (float*)alloc((size_t)TSZ * DM * 4);
  unsigned short* h_bf  = (unsigned short*)alloc((size_t)TSZ * DM * 2);
  unsigned short* h1_bf = (unsigned short*)alloc((size_t)TSZ * 4096 * 2);

  auto cvt = [&](const float* s, unsigned short* dst, int n) {
    f32_to_bf16_kernel<<<(n + 255) / 256, 256, 0, stream>>>(s, dst, n);
  };
  for (int d = 0; d < 2; ++d) {
    cvt(in_w[d],  in_wb[d],  NXZ * DM);
    cvt(xp_w[d],  xp_wb[d],  RCOLS * DI);
    cvt(dt_w[d],  dt_wb[d],  DI * DTRANK);
    cvt(out_w[d], out_wb[d], DM * DI);
  }
  cvt(w1, w1b, 4096 * DM);
  cvt(w2, w2b, DM * 4096);

  ln_kernel<<<TSZ, 256, 0, stream>>>(x, norm_g, norm_b, xn_bf);

  auto tblocks = [](int M, int N) { return (M / BM) * (N / BN); };

  for (int d = 0; d < 2; ++d) {
    // in_proj: xz = xn * in_w^T (bwd uses unflipped xn; flip handled downstream)
    gemm_tiled<0><<<tblocks(TSZ, NXZ), 256, 0, stream>>>(
        xn_bf, DM, in_wb[d], TSZ, NXZ, DM, nullptr, nullptr, xz, nullptr);
    conv_silu_kernel<<<(TSZ * DI + 255) / 256, 256, 0, stream>>>(
        xz, conv_w[d], conv_b[d], u, u_bf, d);
    // x_proj: xdbl = u * xp_w^T (N=96 -> simple kernel)
    gemm_wmma_bf16<2, 0><<<((TSZ / 16) * (RCOLS / 32) + 7) / 8, 256, 0, stream>>>(
        u_bf, DI, xp_wb[d], TSZ, RCOLS, DI, nullptr, nullptr, xdbl, nullptr);
    cvt(xdbl, xdbl_bf, TSZ * RCOLS);
    // dt = softplus(xdbl[:, :64] * dt_w^T + dt_b)  (A lda=96, K=64)
    gemm_tiled<1><<<tblocks(TSZ, DI), 256, 0, stream>>>(
        xdbl_bf, RCOLS, dt_wb[d], TSZ, DI, DTRANK, dt_bias[d], nullptr, dtb, nullptr);
    scan_kernel<<<(NBATCH * DI * 16 + 255) / 256, 256, 0, stream>>>(
        u, dtb, xdbl, A_log[d], yscan);
    gate_kernel<<<(TSZ * DI + 255) / 256, 256, 0, stream>>>(
        yscan, u, xz, Dp[d], y_bf, d);
    // out_proj
    gemm_tiled<0><<<tblocks(TSZ, DM), 256, 0, stream>>>(
        y_bf, DI, out_wb[d], TSZ, DM, DI, nullptr, nullptr, yout[d], nullptr);
  }
  combine_kernel<<<(TSZ * DM + 255) / 256, 256, 0, stream>>>(x, yout[0], yout[1], x2);
  ln_kernel<<<TSZ, 256, 0, stream>>>(x2, ffn_g, ffn_b, h_bf);
  // FFN1: gelu(h*w1^T + b1) -> bf16
  gemm_tiled<2><<<tblocks(TSZ, 4096), 256, 0, stream>>>(
      h_bf, DM, w1b, TSZ, 4096, DM, b1, nullptr, nullptr, h1_bf);
  // FFN2: out = x2 + h1*w2^T + b2
  gemm_tiled<3><<<tblocks(TSZ, DM), 256, 0, stream>>>(
      h1_bf, 4096, w2b, TSZ, DM, 4096, b2, x2, (float*)d_out, nullptr);
}
